// MultiHeadAttention_41738492183140
// MI455X (gfx1250) — matrix operand order repaired and checked by reading, not compiled, not run on hardware
//
#include <hip/hip_runtime.h>

// ---------------------------------------------------------------------------
// Types for CDNA5 WMMA (gfx1250, wave32)
// ---------------------------------------------------------------------------
typedef __attribute__((ext_vector_type(16))) __bf16 v16bf;
typedef __attribute__((ext_vector_type(8)))  __bf16 v8bf;
typedef __attribute__((ext_vector_type(8)))  float  v8f;

union BF16x16 { v16bf v; v8bf h[2]; unsigned u[8]; };
union BF8U    { v8bf v; unsigned short u[8]; };

__device__ __forceinline__ __bf16 f2bf(float f) {
  // round-to-nearest-even fp32 -> bf16
  unsigned int u = __builtin_bit_cast(unsigned int, f);
  u += 0x7FFFu + ((u >> 16) & 1u);
  return __builtin_bit_cast(__bf16, (unsigned short)(u >> 16));
}

// truncating fp32 -> bf16 store (lowers to ds_store_b16_d16_hi: 0 VALU)
__device__ __forceinline__ void store_bf16_trunc(__bf16* p, float f) {
  *(unsigned short*)p =
      (unsigned short)(__builtin_bit_cast(unsigned int, f) >> 16);
}

// low 32 bits of a generic LDS pointer == DS byte address
__device__ __forceinline__ unsigned lds_off(const void* p) {
  return (unsigned)(uintptr_t)p;
}

// CDNA5 async global -> LDS copy (16B per lane), tracked by ASYNCcnt
__device__ __forceinline__ void async_copy_b128(unsigned lds, const void* g) {
  asm volatile("global_load_async_to_lds_b128 %0, %1, off"
               :: "v"(lds), "v"((unsigned long long)(uintptr_t)g)
               : "memory");
}
__device__ __forceinline__ void async_wait0() {
  asm volatile("s_wait_asynccnt 0x0" ::: "memory");
}

// max-reduce across each 16-lane half using DPP16 only (no DS traffic)
template <int CTRL>
__device__ __forceinline__ float dpp_maxstep(float x) {
  int y = __builtin_amdgcn_update_dpp(0, __builtin_bit_cast(int, x),
                                      CTRL, 0xF, 0xF, true);
  return fmaxf(x, __builtin_bit_cast(float, y));
}
__device__ __forceinline__ float maxred16(float x) {
  x = dpp_maxstep<0xB1>(x);   // quad_perm [1,0,3,2]  (xor 1)
  x = dpp_maxstep<0x4E>(x);   // quad_perm [2,3,0,1]  (xor 2)
  x = dpp_maxstep<0x141>(x);  // row_half_mirror      (combine 4s)
  x = dpp_maxstep<0x140>(x);  // row_mirror           (combine 8s)
  return x;
}

// ===========================================================================
// One-time fp32 -> bf16 conversion (pairs, fully coalesced)
// ===========================================================================
__global__ __launch_bounds__(256) void convert_pairs_bf16(
    const float* __restrict__ src, __bf16* __restrict__ dst, int npairs)
{
  int i = blockIdx.x * 256 + threadIdx.x;
  if (i < npairs) {
    float2 f = *(const float2*)(src + (size_t)i * 2);
    unsigned lo = __builtin_bit_cast(unsigned short, f2bf(f.x));
    unsigned hi = __builtin_bit_cast(unsigned short, f2bf(f.y));
    *(unsigned*)(dst + (size_t)i * 2) = lo | (hi << 16);
  }
}

// ===========================================================================
// One-time fp32 [K,N] -> bf16 transposed [N,K] (64x64 LDS tile)
// ===========================================================================
__global__ __launch_bounds__(256) void convert_transpose_bf16(
    const float* __restrict__ src, __bf16* __restrict__ dst, int K, int N)
{
  __shared__ __align__(16) __bf16 T[64 * 72];
  const int t  = threadIdx.x;
  const int n0 = blockIdx.x * 64;
  const int k0 = blockIdx.y * 64;
#pragma unroll
  for (int i = 0; i < 4; ++i) {
    int s = t + i * 256;          // float4 slot (64x64 / 4 = 1024)
    int k = s >> 4, n4 = s & 15;
    float4 f = *(const float4*)(src + (size_t)(k0 + k) * N + n0 + n4 * 4);
    T[(n4 * 4 + 0) * 72 + k] = f2bf(f.x);
    T[(n4 * 4 + 1) * 72 + k] = f2bf(f.y);
    T[(n4 * 4 + 2) * 72 + k] = f2bf(f.z);
    T[(n4 * 4 + 3) * 72 + k] = f2bf(f.w);
  }
  __syncthreads();
#pragma unroll
  for (int i = 0; i < 2; ++i) {
    int s = t + i * 256;          // v8bf slot (64x64 / 8 = 512)
    int n = s >> 3, k8 = s & 7;
    *(v8bf*)(dst + (size_t)(n0 + n) * K + k0 + k8 * 8) =
        *(const v8bf*)(T + n * 72 + k8 * 8);
  }
}

// ===========================================================================
// GEMM + bias: C[M,N] = A[M,K] @ Bt[N,K]^T + bias[N]
// A, Bt bf16 (Bt pre-transposed: staging is a straight async copy).
// Columns n < qcols scaled by qscale (folds softmax scale into Q).
// 128x128 block tile, K-chunk 64, double-buffered async LDS staging:
//   wait(chunk c) -> barrier -> issue async chunk c+1 -> compute chunk c.
// ===========================================================================
template <int N, int K, typename OutT>
__global__ __launch_bounds__(256) void gemm_bias_wmma(
    const __bf16* __restrict__ A, const __bf16* __restrict__ Bt,
    const float* __restrict__ bias, OutT* __restrict__ C,
    float qscale, int qcols)
{
  constexpr int KC  = 64;
  constexpr int BUF = 128 * 72;                  // elems per buffer
  __shared__ __align__(16) __bf16 As[2 * BUF];   // [buf][row][k]
  __shared__ __align__(16) __bf16 Bs[2 * BUF];   // [buf][n][k]

  const int t    = threadIdx.x;
  const int lane = t & 31;
  const int wave = t >> 5;
  const int lm   = lane & 15;
  const int hf   = lane >> 4;
  const int m0   = blockIdx.y * 128;
  const int n0   = blockIdx.x * 128;
  const int wm   = (wave & 3) * 32;
  const int wn   = (wave >> 2) * 64;
  const unsigned asb = lds_off(As);
  const unsigned bsb = lds_off(Bs);

  auto stage = [&](int kk, int bb) {
#pragma unroll
    for (int i = 0; i < 4; ++i) {
      int s   = t + i * 256;      // 1024 16B slots per 128x64 tile
      int row = s >> 3;
      int c8  = s & 7;
      unsigned loff = (unsigned)(bb * BUF + row * 72 + c8 * 8) * 2;
      async_copy_b128(asb + loff, A + (size_t)(m0 + row) * K + kk + c8 * 8);
      async_copy_b128(bsb + loff, Bt + (size_t)(n0 + row) * K + kk + c8 * 8);
    }
  };

  v8f zero = {};
  v8f acc[2][4];
#pragma unroll
  for (int i = 0; i < 2; ++i)
#pragma unroll
    for (int j = 0; j < 4; ++j) acc[i][j] = zero;

  constexpr int NC = K / KC;
  stage(0, 0);
  for (int c = 0; c < NC; ++c) {
    async_wait0();        // own copies of chunk c complete
    __syncthreads();      // => everyone's copies complete, prev reads done
    if (c + 1 < NC) stage((c + 1) * KC, (c + 1) & 1);   // overlap w/ compute

    const __bf16* Ab = As + (c & 1) * BUF;
    const __bf16* Bb = Bs + (c & 1) * BUF;
#pragma unroll
    for (int ks = 0; ks < 2; ++ks) {
      BF16x16 afr[2], bfr[4];
#pragma unroll
      for (int mt = 0; mt < 2; ++mt) {
        int r = wm + mt * 16 + lm;
        // A layout: half0 K{0..7,16..23}, half1 K{8..15,24..31}
        afr[mt].h[0] = *(const v8bf*)(Ab + r * 72 + ks * 32 + 8 * hf);
        afr[mt].h[1] = *(const v8bf*)(Ab + r * 72 + ks * 32 + 16 + 8 * hf);
      }
#pragma unroll
      for (int nt = 0; nt < 4; ++nt) {
        int n = wn + nt * 16 + lm;
        // B layout: half0 K{0..7,16..23}, half1 K{8..15,24..31}
        bfr[nt].h[0] = *(const v8bf*)(Bb + n * 72 + ks * 32 + 8 * hf);
        bfr[nt].h[1] = *(const v8bf*)(Bb + n * 72 + ks * 32 + 16 + 8 * hf);
      }
#pragma unroll
      for (int mt = 0; mt < 2; ++mt)
#pragma unroll
        for (int nt = 0; nt < 4; ++nt)
          acc[mt][nt] = __builtin_amdgcn_wmma_f32_16x16x32_bf16(
              false, afr[mt].v, false, bfr[nt].v, (short)0, acc[mt][nt],
              false, false);
    }
  }

  // ---- bias (+ optional Q scale) + store ----
#pragma unroll
  for (int nt = 0; nt < 4; ++nt) {
    int n    = n0 + wn + nt * 16 + lm;
    float bv = bias[n];
    float sc = (n < qcols) ? qscale : 1.0f;
#pragma unroll
    for (int mt = 0; mt < 2; ++mt) {
#pragma unroll
      for (int r = 0; r < 8; ++r) {
        int m   = m0 + wm + mt * 16 + r + 8 * hf;
        float v = (acc[mt][nt][r] + bv) * sc;
        if constexpr (sizeof(OutT) == 2)
          C[(size_t)m * N + n] = f2bf(v);
        else
          C[(size_t)m * N + n] = v;
      }
    }
  }
}

// ===========================================================================
// Flash attention, bf16 in / bf16 out.  qkv: [B*S, 3E], Q pre-scaled by
// 0.125*log2(e) in GEMM1 epilogue so softmax uses exp2 directly.
// Grid (S/128, B*H); 8 waves x 16 query rows; 64-key tiles, online softmax.
// Double-buffered K (async) and V (split load/store around compute).
// Row max via DPP16; row sum via ones-column WMMA (l shares O's recurrence).
// ===========================================================================
struct VRegs { BF8U r0, r1; };

__global__ __launch_bounds__(256) void flash_attn_bf16_wmma(
    const __bf16* __restrict__ qkv, __bf16* __restrict__ out)
{
  constexpr int S = 2048, E = 1024, LD = 3 * 1024;
  constexpr int KBUF = 64 * 72;

  __shared__ __align__(16) __bf16 Ks[2 * KBUF];   // [buf][key][dh]
  __shared__ __align__(16) __bf16 Vt[2 * KBUF];   // [buf][dh][key]
  __shared__ __align__(16) __bf16 Ps[128 * 72];   // per-wave 16 rows of exp(S)

  const int t    = threadIdx.x;
  const int lane = t & 31;
  const int wave = t >> 5;
  const int lm   = lane & 15;
  const int hf   = lane >> 4;
  const int b    = blockIdx.y >> 4;
  const int h    = blockIdx.y & 15;
  const size_t srow = (size_t)b * S;
  const int q0   = blockIdx.x * 128;
  const unsigned ksb = lds_off(Ks);
  const int vkp  = (t & 31) * 2;    // V staging: even key pair
  const int vdh0 = (t >> 5) * 8;    //            8 dh values

  auto stageK = [&](size_t krow, int bb) {
#pragma unroll
    for (int i = 0; i < 2; ++i) {
      int s   = t + i * 256;        // 512 16B slots (64x64 tile)
      int key = s >> 3;
      int d8  = s & 7;
      async_copy_b128(ksb + (unsigned)(bb * KBUF + key * 72 + d8 * 8) * 2,
                      qkv + (krow + key) * LD + E + h * 64 + d8 * 8);
    }
  };
  auto loadV = [&](size_t krow) {
    const __bf16* vp = qkv + (krow + vkp) * LD + 2 * E + h * 64 + vdh0;
    VRegs v;
    v.r0.v = *(const v8bf*)(vp);
    v.r1.v = *(const v8bf*)(vp + LD);
    return v;
  };
  auto storeV = [&](VRegs v, int bb) {
    __bf16* dst = Vt + bb * KBUF;
#pragma unroll
    for (int e = 0; e < 8; ++e) {
      unsigned pk = (unsigned)v.r0.u[e] | ((unsigned)v.r1.u[e] << 16);
      *(unsigned*)(dst + (vdh0 + e) * 72 + vkp) = pk;
    }
  };

  // ---- load Q fragments once (already scaled) ----
  BF16x16 qf[2];
  {
    const __bf16* qp = qkv + (srow + q0 + wave * 16 + lm) * LD + h * 64;
#pragma unroll
    for (int ks = 0; ks < 2; ++ks) {
      qf[ks].h[0] = *(const v8bf*)(qp + ks * 32 + 8 * hf);
      qf[ks].h[1] = *(const v8bf*)(qp + ks * 32 + 16 + 8 * hf);
    }
  }

  // ones B-fragment: every column = 1.0bf16 -> P@ones puts the row sum
  // in every lane of the C fragment (no broadcast needed, ever).
  BF16x16 ones;
#pragma unroll
  for (int i = 0; i < 8; ++i) ones.u[i] = 0x3F803F80u;

  float mrow[8];
  v8f zero = {};
  v8f oacc[4], osum = zero;
#pragma unroll
  for (int r = 0; r < 8; ++r) mrow[r] = -3.0e38f;
#pragma unroll
  for (int nt = 0; nt < 4; ++nt) oacc[nt] = zero;

  constexpr int NJ = S / 64;
  stageK(srow, 0);
  storeV(loadV(srow), 0);

  for (int j = 0; j < NJ; ++j) {
    async_wait0();
    __syncthreads();

    const bool more = (j + 1 < NJ);
    VRegs vn = {};
    if (more) {
      stageK(srow + (size_t)(j + 1) * 64, (j + 1) & 1);  // async, overlapped
      vn = loadV(srow + (size_t)(j + 1) * 64);           // issue loads early
    }
    const __bf16* Kb = Ks + (j & 1) * KBUF;
    const __bf16* Vb = Vt + (j & 1) * KBUF;

    // ---- scores: S = Q @ K^T (16 x 64 per wave) ----
    v8f sf[4];
#pragma unroll
    for (int nt = 0; nt < 4; ++nt) {
      v8f sc = zero;
#pragma unroll
      for (int ks = 0; ks < 2; ++ks) {
        BF16x16 bf_;
        const int key = nt * 16 + lm;
        bf_.h[0] = *(const v8bf*)(Kb + key * 72 + ks * 32 + 8 * hf);
        bf_.h[1] = *(const v8bf*)(Kb + key * 72 + ks * 32 + 16 + 8 * hf);
        sc = __builtin_amdgcn_wmma_f32_16x16x32_bf16(
            false, qf[ks].v, false, bf_.v, (short)0, sc, false, false);
      }
      sf[nt] = sc;
    }

    // ---- online softmax (base-2): row max via DPP16, VALU-only ----
#pragma unroll
    for (int r = 0; r < 8; ++r) {
      float x = fmaxf(fmaxf(sf[0][r], sf[1][r]), fmaxf(sf[2][r], sf[3][r]));
      x = maxred16(x);
      float mnew = fmaxf(mrow[r], x);
      float corr = exp2f(mrow[r] - mnew);
      mrow[r] = mnew;
      osum[r] *= corr;
      oacc[0][r] *= corr; oacc[1][r] *= corr;
      oacc[2][r] *= corr; oacc[3][r] *= corr;
    }

    // ---- P = exp2(S - m) -> LDS (truncating bf16) ----
#pragma unroll
    for (int nt = 0; nt < 4; ++nt) {
#pragma unroll
      for (int r = 0; r < 8; ++r) {
        float p = exp2f(sf[nt][r] - mrow[r]);
        store_bf16_trunc(&Ps[(wave * 16 + r + 8 * hf) * 72 + nt * 16 + lm], p);
      }
    }

    // ---- O += P @ V ;  l += P @ ones (same recurrence as O) ----
#pragma unroll
    for (int ks = 0; ks < 2; ++ks) {
      BF16x16 pa;
      const int pr = wave * 16 + lm;
      pa.h[0] = *(const v8bf*)(Ps + pr * 72 + ks * 32 + 8 * hf);
      pa.h[1] = *(const v8bf*)(Ps + pr * 72 + ks * 32 + 16 + 8 * hf);
#pragma unroll
      for (int nt = 0; nt < 4; ++nt) {
        BF16x16 vb;
        const int dh = nt * 16 + lm;
        vb.h[0] = *(const v8bf*)(Vb + dh * 72 + ks * 32 + 8 * hf);
        vb.h[1] = *(const v8bf*)(Vb + dh * 72 + ks * 32 + 16 + 8 * hf);
        oacc[nt] = __builtin_amdgcn_wmma_f32_16x16x32_bf16(
            false, pa.v, false, vb.v, (short)0, oacc[nt], false, false);
      }
      osum = __builtin_amdgcn_wmma_f32_16x16x32_bf16(
          false, pa.v, false, ones.v, (short)0, osum, false, false);
    }

    // ---- finish V staging for next tile (loads have landed by now) ----
    if (more) storeV(vn, (j + 1) & 1);
  }

  // ---- normalize + store bf16 (every lane already holds its row sums) ----
  float inv[8];
#pragma unroll
  for (int r = 0; r < 8; ++r) inv[r] = 1.0f / osum[r];
#pragma unroll
  for (int nt = 0; nt < 4; ++nt) {
#pragma unroll
    for (int r = 0; r < 8; ++r) {
      size_t row = srow + q0 + wave * 16 + r + 8 * hf;
      out[row * E + h * 64 + nt * 16 + lm] = f2bf(oacc[nt][r] * inv[r]);
    }
  }
}

// ===========================================================================
// Launch
// ===========================================================================
extern "C" void kernel_launch(void* const* d_in, const int* in_sizes, int n_in,
                              void* d_out, int out_size, void* d_ws, size_t ws_size,
                              hipStream_t stream) {
  (void)in_sizes; (void)n_in; (void)out_size; (void)ws_size;
  const float* x      = (const float*)d_in[0];   // [4096,1024]
  const float* w_qkv  = (const float*)d_in[1];   // [1024,3072]
  const float* b_qkv  = (const float*)d_in[2];   // [3072]
  const float* w_proj = (const float*)d_in[3];   // [1024,1024]
  const float* b_proj = (const float*)d_in[4];   // [1024]
  float* out = (float*)d_out;                    // [4096,1024]

  const int M = 4096;
  __bf16* x_bf    = (__bf16*)d_ws;                       // [4096,1024]
  __bf16* wqkvT   = x_bf   + (size_t)M * 1024;           // [3072,1024]
  __bf16* wprojT  = wqkvT  + (size_t)3072 * 1024;        // [1024,1024]
  __bf16* qkv_bf  = wprojT + (size_t)1024 * 1024;        // [4096,3072]
  __bf16* attn_bf = qkv_bf + (size_t)M * 3072;           // [4096,1024]

  // 0) one-time bf16 conversions (+ weight transposes for copy-only staging)
  convert_pairs_bf16<<<(M * 1024 / 2 + 255) / 256, 256, 0, stream>>>(
      x, x_bf, M * 1024 / 2);
  convert_transpose_bf16<<<dim3(3072 / 64, 1024 / 64), 256, 0, stream>>>(
      w_qkv, wqkvT, 1024, 3072);
  convert_transpose_bf16<<<dim3(1024 / 64, 1024 / 64), 256, 0, stream>>>(
      w_proj, wprojT, 1024, 1024);

  // 1) QKV projection (Q columns pre-scaled by 0.125*log2(e))
  gemm_bias_wmma<3072, 1024, __bf16><<<dim3(24, M / 128), 256, 0, stream>>>(
      x_bf, wqkvT, b_qkv, qkv_bf, 0.125f * 1.44269504f, 1024);

  // 2) Flash attention
  flash_attn_bf16_wmma<<<dim3(16, 32), 256, 0, stream>>>(qkv_bf, attn_bf);

  // 3) Output projection (fp32 out)
  gemm_bias_wmma<1024, 1024, float><<<dim3(8, M / 128), 256, 0, stream>>>(
      attn_bf, wprojT, b_proj, out, 1.0f, 0);
}
